// EnhancedFusionModel_71184787964137
// MI455X (gfx1250) — matrix operand order repaired and checked by reading, not compiled, not run on hardware
//
#include <hip/hip_runtime.h>
#include <hip/hip_bf16.h>
#include <math.h>

// ---------------------------------------------------------------------------
// EnhancedFusionModel forward for MI455X (gfx1250, wave32, WMMA bf16).
// GEMMs run on v_wmma_f32_16x16x32_bf16 with fp32 accumulation; LDS tiles are
// stored in *fragment-native* order so operand loads are 2x ds_load_b128 per
// fragment (no scalar gather). Weights are pre-packed once into this order.
// MHA(seq=1) is pre-composed to a single 64x64 GEMM; nn.Bilinear becomes a
// reshaped GEMM + per-row rank reduction. LayerNorm is done in-register with
// wave32 butterfly shuffles.
// ---------------------------------------------------------------------------

#define DEV __device__ __forceinline__

typedef __attribute__((ext_vector_type(16))) __bf16 v16bf;
typedef __attribute__((ext_vector_type(8)))  float  v8f;

DEV unsigned short f2bf(float f) {
  unsigned u = __builtin_bit_cast(unsigned, f);
  unsigned r = u + 0x7FFFu + ((u >> 16) & 1u);   // round-to-nearest-even
  return (unsigned short)(r >> 16);
}
DEV float  bfus2f(unsigned short u) { unsigned x = ((unsigned)u) << 16; return __builtin_bit_cast(float, x); }
DEV float  gelu_erf(float x) { return 0.5f * x * (1.0f + erff(x * 0.70710678118654752440f)); }
DEV int    gate_seg(int k) { return k < 192 ? (k >> 6) : 3 + ((k - 192) >> 5); }

// Packed bf16 weight layout, per-lane B-operand fragment order used by this code (contiguous k, same as A):
// for (k, c) of a [K, N] matrix: kt=k>>5, e=k&15, lane=(c&15)+16*((k>>4)&1),
// nt=c>>4 -> offset ((kt*NTtot + nt)*32 + lane)*16 + e.
DEV size_t packed_off(int k, int c, int NTtot) {
  int kt = k >> 5;
  int nt = c >> 4;
  int l  = (c & 15) + (((k >> 4) & 1) << 4);
  int e  = k & 15;
  return ((((size_t)kt * NTtot + nt) * 32) + l) * 16 + e;
}

// --------------------------------- GEMM ------------------------------------
enum { EPI_RB1 = 0, EPI_RB2 = 1, EPI_MHA = 2, EPI_GELU_LN = 3, EPI_Y = 4 };

struct GemmArgs {
  const float* A;  const float* A2;          // fp32 activations
  const unsigned short* W; const unsigned short* W2;  // packed bf16 weights
  int K, K2, lda, lda2, ntot;                // ntot = total N tiles of W
  const float* bias;                         // [N]
  const float* gamma; const float* beta;     // [16*LT]
  const float* res; int ldres;               // residual / base tensor
  float* outF; int ldout;                    // fp32 out
  float* outR; int ldoutR;                   // RB1 residual half out
  unsigned short* outB; int ldoutB;          // bf16 out (Y)
  const float* gate;                         // [M x 6]
};

template <int EPI, int NT, int LT, bool DUAL, bool GATE>
__global__ __launch_bounds__(256)
void wmma_gemm(GemmArgs g) {
  constexpr int N = NT * 16;
  // fragment-native tiles: per lane 16 contiguous bf16 (32B vector loads)
  __shared__ __align__(16) unsigned short Af[8 * 32 * 16];    // 8 waves A frags
  __shared__ __align__(16) unsigned short Bf[NT * 32 * 16];   // NT B frags

  const int tid = threadIdx.x;
  const int wave = tid >> 5, lane = tid & 31;
  const int l16 = lane & 15, half = lane >> 4;
  const long rowBase = (long)blockIdx.x * 128;

  int ntBase = 0;
  unsigned short* outBb = g.outB;
  if constexpr (EPI == EPI_Y) {               // column-block tiling over grid.y
    ntBase = blockIdx.y * NT;
    outBb += (size_t)blockIdx.y * N;
  }

  const v8f vzero = {0.f, 0.f, 0.f, 0.f, 0.f, 0.f, 0.f, 0.f};
  v8f acc[NT];
#pragma unroll
  for (int nt = 0; nt < NT; ++nt) acc[nt] = vzero;

  const int npass = DUAL ? 2 : 1;
  for (int pass = 0; pass < npass; ++pass) {
    const float* A = pass ? g.A2 : g.A;
    const unsigned short* W = pass ? g.W2 : g.W;
    const int K   = pass ? g.K2 : g.K;
    const int lda = pass ? g.lda2 : g.lda;

    for (int k0 = 0; k0 < K; k0 += 32) {
      __syncthreads();                        // previous tile fully consumed

      // ---- stage A tile 128x32 fp32 -> bf16, fragment order ----
      // Each thread: one float4 per sweep; 4-aligned column groups never
      // cross the half/e boundaries, so the 4 bf16 land contiguously.
#pragma unroll
      for (int i = 0; i < 4; ++i) {
        int idx4 = (tid + i * 256) << 2;      // linear element index
        int r = idx4 >> 5, c = idx4 & 31;     // row 0..127, col 0,4,..,28
        const float* ap = &A[(rowBase + r) * (long)lda + (k0 + c)];
        float4 av = *(const float4*)ap;
        if constexpr (GATE) {
          float gs = g.gate[(rowBase + r) * 6 + gate_seg(k0 + c)];
          av.x *= gs; av.y *= gs; av.z *= gs; av.w *= gs;
        }
        if (k0 + 32 < K) __builtin_prefetch(ap + 32, 0, 1);
        int w = r >> 4, m = r & 15;
        int h = c >> 4;
        int e0 = c & 15;
        int l = m + (h << 4);
        unsigned short p[4] = {f2bf(av.x), f2bf(av.y), f2bf(av.z), f2bf(av.w)};
        *(uint2*)&Af[((w * 32 + l) * 16 + e0)] = *(const uint2*)p;
      }
      // ---- stage B tile: straight linear copy of packed weights ----
      {
        const uint4* src = (const uint4*)(W + (((size_t)(k0 >> 5) * g.ntot + ntBase) * 512));
        uint4* dst = (uint4*)Bf;
#pragma unroll
        for (int i = tid; i < NT * 64; i += 256) dst[i] = src[i];
      }
      __syncthreads();

      const v16bf af = *(const v16bf*)&Af[(wave * 32 + lane) * 16];
#pragma unroll
      for (int nt = 0; nt < NT; ++nt) {
        const v16bf bfv = *(const v16bf*)&Bf[(nt * 32 + lane) * 16];
        acc[nt] = __builtin_amdgcn_wmma_f32_16x16x32_bf16(
            false, af, false, bfv, (short)0, acc[nt], false, false);
      }
    }
  }

  // ------------------------------ epilogue ---------------------------------
  const long rrow0 = rowBase + wave * 16 + (half << 3);   // + r = global row

  if constexpr (EPI == EPI_Y) {
#pragma unroll
    for (int nt = 0; nt < NT; ++nt) {
      int col = nt * 16 + l16;
#pragma unroll
      for (int r = 0; r < 8; ++r)
        outBb[(rrow0 + r) * (size_t)g.ldoutB + col] = f2bf(acc[nt][r]);
    }
  } else {
    // pre-transform
    if constexpr (EPI == EPI_RB1 || EPI == EPI_RB2) {
#pragma unroll
      for (int nt = 0; nt < NT; ++nt) {
        float b = g.bias[nt * 16 + l16];
#pragma unroll
        for (int r = 0; r < 8; ++r) acc[nt][r] += b;
      }
    } else if constexpr (EPI == EPI_MHA) {
#pragma unroll
      for (int nt = 0; nt < NT; ++nt) {
        int col = nt * 16 + l16;
        float b = g.bias[col];
#pragma unroll
        for (int r = 0; r < 8; ++r)
          acc[nt][r] = g.res[(rrow0 + r) * (size_t)g.ldres + col] + 0.5f * acc[nt][r] + b;
      }
    } else if constexpr (EPI == EPI_GELU_LN) {
#pragma unroll
      for (int nt = 0; nt < NT; ++nt) {
        float b = g.bias[nt * 16 + l16];
#pragma unroll
        for (int r = 0; r < 8; ++r) acc[nt][r] = gelu_erf(acc[nt][r] + b);
      }
    }

    // LayerNorm over first LT tiles (per-row; wave owns all N columns).
    float mean[8], rstd[8];
#pragma unroll
    for (int r = 0; r < 8; ++r) {
      float s = 0.f, q = 0.f;
#pragma unroll
      for (int nt = 0; nt < LT; ++nt) { float v = acc[nt][r]; s += v; q += v * v; }
#pragma unroll
      for (int m = 1; m <= 8; m <<= 1) {      // butterfly within 16-lane half
        s += __shfl_xor(s, m, 32);
        q += __shfl_xor(q, m, 32);
      }
      float inv = 1.0f / (LT * 16);
      float mu = s * inv, var = q * inv - mu * mu;
      mean[r] = mu;
      rstd[r] = rsqrtf(var + 1e-5f);
    }

#pragma unroll
    for (int nt = 0; nt < NT; ++nt) {
      int col = nt * 16 + l16;
      if (nt < LT) {
        float gm = g.gamma[col], bt = g.beta[col];
#pragma unroll
        for (int r = 0; r < 8; ++r) {
          float v = (acc[nt][r] - mean[r]) * rstd[r] * gm + bt;
          long row = rrow0 + r;
          if constexpr (EPI == EPI_RB1)       v = gelu_erf(v);
          else if constexpr (EPI == EPI_RB2)  v = gelu_erf(v + g.res[row * (size_t)g.ldres + col]);
          g.outF[row * (size_t)g.ldout + col] = v;
        }
      } else {                                // RB1 residual half (raw)
#pragma unroll
        for (int r = 0; r < 8; ++r)
          g.outR[(rrow0 + r) * (size_t)g.ldoutR + (col - LT * 16)] = acc[nt][r];
      }
    }
  }
}

// ------------------------------ prep kernels -------------------------------
// [w1|ws] -> packed bf16 (fragment order), b1|bs -> concatenated f32 bias
__global__ void k_pack_rb(const float* w1, const float* ws, const float* b1, const float* bs,
                          int K, int dout, unsigned short* Wd, float* bd) {
  int n2 = 2 * dout;
  long total = (long)K * n2;
  for (long idx = blockIdx.x * (long)blockDim.x + threadIdx.x; idx < total;
       idx += (long)gridDim.x * blockDim.x) {
    int k = (int)(idx / n2), c = (int)(idx % n2);
    float v = (c < dout) ? w1[(long)k * dout + c] : ws[(long)k * dout + (c - dout)];
    Wd[packed_off(k, c, n2 >> 4)] = f2bf(v);
    if (idx < n2) bd[idx] = (idx < dout) ? b1[idx] : bs[idx - dout];
  }
}

// generic [K,N] f32 -> packed bf16
__global__ void k_pack_w(const float* src, int K, int N, unsigned short* dst) {
  long total = (long)K * N;
  for (long idx = blockIdx.x * (long)blockDim.x + threadIdx.x; idx < total;
       idx += (long)gridDim.x * blockDim.x) {
    int k = (int)(idx / N), c = (int)(idx % N);
    dst[packed_off(k, c, N >> 4)] = f2bf(src[idx]);
  }
}

// Compose W' = wv@wo (packed bf16) for a pair of MHAs + bsum = 0.5*(bv@wo+bo)
__global__ void k_mha_pair(const float* wvA, const float* bvA, const float* woA, const float* boA,
                           const float* wvB, const float* bvB, const float* woB, const float* boB,
                           unsigned short* WA, unsigned short* WB, float* bsum) {
  for (int idx = blockIdx.x * blockDim.x + threadIdx.x; idx < 8192;
       idx += gridDim.x * blockDim.x) {
    int which = idx >> 12;
    int ij = idx & 4095;
    int i = ij >> 6, j = ij & 63;
    const float* wv = which ? wvB : wvA;
    const float* wo = which ? woB : woA;
    float s = 0.f;
    for (int k = 0; k < 64; ++k) s += wv[i * 64 + k] * wo[k * 64 + j];
    (which ? WB : WA)[packed_off(i, j, 4)] = f2bf(s);
    if (idx < 64) {
      float sa = 0.f, sb = 0.f;
      for (int k = 0; k < 64; ++k) {
        sa += bvA[k] * woA[k * 64 + idx];
        sb += bvB[k] * woB[k * 64 + idx];
      }
      bsum[idx] = 0.5f * (sa + boA[idx] + sb + boB[idx]);
    }
  }
}

// Bilinear W [32,64,64] -> packed bf16 [64, 2048]: (k=i, c=o*64+j) = W[o,i,j]
__global__ void k_pack_bl(const float* W, unsigned short* Wr) {
  for (int idx = blockIdx.x * blockDim.x + threadIdx.x; idx < 64 * 2048;
       idx += gridDim.x * blockDim.x) {
    int i = idx >> 11;
    int rem = idx & 2047;
    int o = rem >> 6, j = rem & 63;
    Wr[packed_off(i, o * 64 + j, 128)] = f2bf(W[o * 4096 + i * 64 + j]);
  }
}

// ------------------------------ VALU kernels -------------------------------
// ti[b,o] = sum_j Y[b,o*64+j]*x2[b,j] + bias[o]; LN over 32; gelu -> comb col
__global__ void k_bilin_reduce(const unsigned short* Y, int rows, float* comb,
                               int x2off, int outoff, const float* bias,
                               const float* gamma, const float* beta) {
  int wave = threadIdx.x >> 5, lane = threadIdx.x & 31;
  for (int row = blockIdx.x * 8 + wave; row < rows; row += gridDim.x * 8) {
    const ushort4* y4 = (const ushort4*)(Y + (size_t)row * 2048 + lane * 64);
    const float4* x4 = (const float4*)(comb + (size_t)row * 288 + x2off);
    float s = 0.f;
#pragma unroll
    for (int j = 0; j < 16; ++j) {
      ushort4 yv = y4[j];
      float4  xv = x4[j];
      s += bfus2f(yv.x) * xv.x + bfus2f(yv.y) * xv.y +
           bfus2f(yv.z) * xv.z + bfus2f(yv.w) * xv.w;
    }
    s += bias[lane];
    float sum = s, sq = s * s;
#pragma unroll
    for (int m = 1; m <= 16; m <<= 1) {
      sum += __shfl_xor(sum, m, 32);
      sq  += __shfl_xor(sq,  m, 32);
    }
    float mu = sum * (1.f / 32.f), var = sq * (1.f / 32.f) - mu * mu;
    float v = (s - mu) * rsqrtf(var + 1e-5f) * gamma[lane] + beta[lane];
    comb[(size_t)row * 288 + outoff + lane] = gelu_erf(v);
  }
}

__global__ void k_gate(const float* comb, const float* wg, const float* bg,
                       float* gate, int rows) {
  int wave = threadIdx.x >> 5, lane = threadIdx.x & 31;
  for (int row = blockIdx.x * 8 + wave; row < rows; row += gridDim.x * 8) {
    float p0 = 0, p1 = 0, p2 = 0, p3 = 0, p4 = 0, p5 = 0;
    for (int k = lane; k < 288; k += 32) {
      float x = comb[(size_t)row * 288 + k];
      const float* w = wg + k * 6;
      p0 += x * w[0]; p1 += x * w[1]; p2 += x * w[2];
      p3 += x * w[3]; p4 += x * w[4]; p5 += x * w[5];
    }
#pragma unroll
    for (int m = 1; m <= 16; m <<= 1) {
      p0 += __shfl_xor(p0, m, 32); p1 += __shfl_xor(p1, m, 32);
      p2 += __shfl_xor(p2, m, 32); p3 += __shfl_xor(p3, m, 32);
      p4 += __shfl_xor(p4, m, 32); p5 += __shfl_xor(p5, m, 32);
    }
    if (lane == 0) {
      float p[6] = {p0, p1, p2, p3, p4, p5};
      for (int o = 0; o < 6; ++o)
        gate[(size_t)row * 6 + o] = 1.f / (1.f + expf(-(p[o] + bg[o])));
    }
  }
}

__global__ void k_final(const float* feat, const float* wf2, const float* bf2,
                        float* out, int rows) {
  int wave = threadIdx.x >> 5, lane = threadIdx.x & 31;
  for (int row = blockIdx.x * 8 + wave; row < rows; row += gridDim.x * 8) {
    float s = feat[(size_t)row * 64 + lane] * wf2[lane] +
              feat[(size_t)row * 64 + 32 + lane] * wf2[32 + lane];
#pragma unroll
    for (int m = 1; m <= 16; m <<= 1) s += __shfl_xor(s, m, 32);
    if (lane == 0) out[row] = s + bf2[0];
  }
}

// ------------------------------ host driver --------------------------------
// d_in flattening assumption: recursive dict-insertion order of setup_inputs():
// 0 tab, 1 img, 2 text, then params as written in _make_params (RB leaves in
// order w1,b1,w2,b2,g1,be1,g2,be2,ws,bs).
extern "C" void kernel_launch(void* const* d_in, const int* in_sizes, int n_in,
                              void* d_out, int out_size, void* d_ws, size_t ws_size,
                              hipStream_t stream) {
  (void)in_sizes; (void)out_size; (void)ws_size;
  if (n_in < 123) return;
  const int B = 32768;
  const int CHUNK = 4096;

  // param indices
  const int P_TAB1 = 3, P_TAB2 = 13, P_IMG1 = 23, P_IMG2 = 33, P_TXT1 = 43, P_TXT2 = 53;
  const int P_T2I = 63, P_T2X = 67, P_I2T = 71, P_I2X = 75, P_X2T = 79, P_X2I = 83;
  const int P_NG_TAB = 87, P_NG_IMG = 89, P_NG_TXT = 91;
  const int P_BL_TI = 93, P_BL_TX = 97, P_BL_IX = 101;
  const int P_WG = 105, P_BG = 106, P_WF1 = 107, P_BF1 = 108, P_NGF = 109, P_NBF = 110;
  const int P_RBF = 111, P_WF2 = 121, P_BF2 = 122;

  char* ws = (char*)d_ws;
  size_t off = 0;
  auto alloc = [&](size_t bytes) { size_t o = off; off = (off + bytes + 255) & ~(size_t)255; return o; };

  // packed bf16 weight regions
  size_t WC_TAB1 = alloc(256 * 256 * 2), WC_IMG1 = alloc(2048 * 256 * 2), WC_TXT1 = alloc(768 * 256 * 2);
  size_t W2_TAB1 = alloc(128 * 128 * 2), W2_IMG1 = alloc(128 * 128 * 2), W2_TXT1 = alloc(128 * 128 * 2);
  size_t WC_TAB2 = alloc(128 * 128 * 2), WC_IMG2 = alloc(128 * 128 * 2), WC_TXT2 = alloc(128 * 128 * 2);
  size_t W2_TAB2 = alloc(64 * 64 * 2),  W2_IMG2 = alloc(64 * 64 * 2),  W2_TXT2 = alloc(64 * 64 * 2);
  size_t WC_RBF  = alloc(128 * 128 * 2), W2_RBF = alloc(64 * 64 * 2);
  size_t WM_T2I = alloc(4096 * 2), WM_T2X = alloc(4096 * 2), WM_I2T = alloc(4096 * 2);
  size_t WM_I2X = alloc(4096 * 2), WM_X2T = alloc(4096 * 2), WM_X2I = alloc(4096 * 2);
  size_t WF1B = alloc(288 * 128 * 2);
  size_t WBL_TI = alloc(64 * 2048 * 2), WBL_TX = alloc(64 * 2048 * 2), WBL_IX = alloc(64 * 2048 * 2);
  // fp32 bias cats / bsums
  size_t BC_TAB1 = alloc(256 * 4), BC_IMG1 = alloc(256 * 4), BC_TXT1 = alloc(256 * 4);
  size_t BC_TAB2 = alloc(128 * 4), BC_IMG2 = alloc(128 * 4), BC_TXT2 = alloc(128 * 4), BC_RBF = alloc(128 * 4);
  size_t BSUM_TAB = alloc(64 * 4), BSUM_IMG = alloc(64 * 4), BSUM_TXT = alloc(64 * 4);
  // activations
  size_t H128 = alloc((size_t)B * 128 * 4), R128 = alloc((size_t)B * 128 * 4), F128 = alloc((size_t)B * 128 * 4);
  size_t H64 = alloc((size_t)B * 64 * 4), R64 = alloc((size_t)B * 64 * 4);
  size_t FT = alloc((size_t)B * 64 * 4), FI = alloc((size_t)B * 64 * 4), FX = alloc((size_t)B * 64 * 4);
  size_t COMB = alloc((size_t)B * 288 * 4);
  size_t GATE = alloc((size_t)B * 6 * 4);
  size_t YC = alloc((size_t)CHUNK * 2048 * 2);

  auto F  = [&](int i) { return (const float*)d_in[i]; };
  auto US = [&](size_t o) { return (unsigned short*)(ws + o); };
  auto FP = [&](size_t o) { return (float*)(ws + o); };

  dim3 T(256);

  // ---- prep: pack RB weights (w1|ws -> packed bf16, b1|bs cat) ----
  k_pack_rb<<<dim3(512), T, 0, stream>>>(F(P_TAB1+0), F(P_TAB1+8), F(P_TAB1+1), F(P_TAB1+9), 256, 128, US(WC_TAB1), FP(BC_TAB1));
  k_pack_rb<<<dim3(512), T, 0, stream>>>(F(P_IMG1+0), F(P_IMG1+8), F(P_IMG1+1), F(P_IMG1+9), 2048, 128, US(WC_IMG1), FP(BC_IMG1));
  k_pack_rb<<<dim3(512), T, 0, stream>>>(F(P_TXT1+0), F(P_TXT1+8), F(P_TXT1+1), F(P_TXT1+9), 768, 128, US(WC_TXT1), FP(BC_TXT1));
  k_pack_rb<<<dim3(256), T, 0, stream>>>(F(P_TAB2+0), F(P_TAB2+8), F(P_TAB2+1), F(P_TAB2+9), 128, 64, US(WC_TAB2), FP(BC_TAB2));
  k_pack_rb<<<dim3(256), T, 0, stream>>>(F(P_IMG2+0), F(P_IMG2+8), F(P_IMG2+1), F(P_IMG2+9), 128, 64, US(WC_IMG2), FP(BC_IMG2));
  k_pack_rb<<<dim3(256), T, 0, stream>>>(F(P_TXT2+0), F(P_TXT2+8), F(P_TXT2+1), F(P_TXT2+9), 128, 64, US(WC_TXT2), FP(BC_TXT2));
  k_pack_rb<<<dim3(256), T, 0, stream>>>(F(P_RBF+0),  F(P_RBF+8),  F(P_RBF+1),  F(P_RBF+9),  128, 64, US(WC_RBF),  FP(BC_RBF));
  // ---- prep: pack w2 matrices + wf1 ----
  k_pack_w<<<dim3(64), T, 0, stream>>>(F(P_TAB1+2), 128, 128, US(W2_TAB1));
  k_pack_w<<<dim3(64), T, 0, stream>>>(F(P_IMG1+2), 128, 128, US(W2_IMG1));
  k_pack_w<<<dim3(64), T, 0, stream>>>(F(P_TXT1+2), 128, 128, US(W2_TXT1));
  k_pack_w<<<dim3(64), T, 0, stream>>>(F(P_TAB2+2), 64, 64, US(W2_TAB2));
  k_pack_w<<<dim3(64), T, 0, stream>>>(F(P_IMG2+2), 64, 64, US(W2_IMG2));
  k_pack_w<<<dim3(64), T, 0, stream>>>(F(P_TXT2+2), 64, 64, US(W2_TXT2));
  k_pack_w<<<dim3(64), T, 0, stream>>>(F(P_RBF+2),  64, 64, US(W2_RBF));
  k_pack_w<<<dim3(64), T, 0, stream>>>(F(P_WF1),    288, 128, US(WF1B));
  // ---- prep: compose MHA weight products ----
  k_mha_pair<<<dim3(64), T, 0, stream>>>(F(P_T2I+0), F(P_T2I+1), F(P_T2I+2), F(P_T2I+3),
                                         F(P_T2X+0), F(P_T2X+1), F(P_T2X+2), F(P_T2X+3),
                                         US(WM_T2I), US(WM_T2X), FP(BSUM_TAB));
  k_mha_pair<<<dim3(64), T, 0, stream>>>(F(P_I2T+0), F(P_I2T+1), F(P_I2T+2), F(P_I2T+3),
                                         F(P_I2X+0), F(P_I2X+1), F(P_I2X+2), F(P_I2X+3),
                                         US(WM_I2T), US(WM_I2X), FP(BSUM_IMG));
  k_mha_pair<<<dim3(64), T, 0, stream>>>(F(P_X2T+0), F(P_X2T+1), F(P_X2T+2), F(P_X2T+3),
                                         F(P_X2I+0), F(P_X2I+1), F(P_X2I+2), F(P_X2I+3),
                                         US(WM_X2T), US(WM_X2I), FP(BSUM_TXT));
  // ---- prep: permute bilinear weights ----
  k_pack_bl<<<dim3(256), T, 0, stream>>>(F(P_BL_TI), US(WBL_TI));
  k_pack_bl<<<dim3(256), T, 0, stream>>>(F(P_BL_TX), US(WBL_TX));
  k_pack_bl<<<dim3(256), T, 0, stream>>>(F(P_BL_IX), US(WBL_IX));

  const dim3 G(B / 128);

  // ---- modality towers ----
  struct TW { const float* x; int Kin; int p1, p2; size_t wc1, w21, bc1, wc2, w22, bc2, feat; };
  TW tw[3] = {
    {F(0), 256,  P_TAB1, P_TAB2, WC_TAB1, W2_TAB1, BC_TAB1, WC_TAB2, W2_TAB2, BC_TAB2, FT},
    {F(1), 2048, P_IMG1, P_IMG2, WC_IMG1, W2_IMG1, BC_IMG1, WC_IMG2, W2_IMG2, BC_IMG2, FI},
    {F(2), 768,  P_TXT1, P_TXT2, WC_TXT1, W2_TXT1, BC_TXT1, WC_TXT2, W2_TXT2, BC_TXT2, FX},
  };
  for (int t = 0; t < 3; ++t) {
    { // RB1 part1: [w1|ws] -> H128, R128
      GemmArgs a{}; a.A = tw[t].x; a.lda = tw[t].Kin; a.K = tw[t].Kin;
      a.W = US(tw[t].wc1); a.ntot = 16; a.bias = FP(tw[t].bc1);
      a.gamma = F(tw[t].p1 + 4); a.beta = F(tw[t].p1 + 5);
      a.outF = FP(H128); a.ldout = 128; a.outR = FP(R128); a.ldoutR = 128;
      wmma_gemm<EPI_RB1, 16, 8, false, false><<<G, T, 0, stream>>>(a);
    }
    { // RB1 part2: gelu(LN(H@w2+b2)+R) -> F128
      GemmArgs a{}; a.A = FP(H128); a.lda = 128; a.K = 128;
      a.W = US(tw[t].w21); a.ntot = 8; a.bias = F(tw[t].p1 + 3);
      a.gamma = F(tw[t].p1 + 6); a.beta = F(tw[t].p1 + 7);
      a.res = FP(R128); a.ldres = 128; a.outF = FP(F128); a.ldout = 128;
      wmma_gemm<EPI_RB2, 8, 8, false, false><<<G, T, 0, stream>>>(a);
    }
    { // RB2 part1 -> H64, R64
      GemmArgs a{}; a.A = FP(F128); a.lda = 128; a.K = 128;
      a.W = US(tw[t].wc2); a.ntot = 8; a.bias = FP(tw[t].bc2);
      a.gamma = F(tw[t].p2 + 4); a.beta = F(tw[t].p2 + 5);
      a.outF = FP(H64); a.ldout = 64; a.outR = FP(R64); a.ldoutR = 64;
      wmma_gemm<EPI_RB1, 8, 4, false, false><<<G, T, 0, stream>>>(a);
    }
    { // RB2 part2 -> feat
      GemmArgs a{}; a.A = FP(H64); a.lda = 64; a.K = 64;
      a.W = US(tw[t].w22); a.ntot = 4; a.bias = F(tw[t].p2 + 3);
      a.gamma = F(tw[t].p2 + 6); a.beta = F(tw[t].p2 + 7);
      a.res = FP(R64); a.ldres = 64; a.outF = FP(tw[t].feat); a.ldout = 64;
      wmma_gemm<EPI_RB2, 4, 4, false, false><<<G, T, 0, stream>>>(a);
    }
  }

  // ---- cross-attention fusion (dual-A accumulating GEMMs) -> comb cols ----
  { GemmArgs a{}; a.A = FP(FI); a.A2 = FP(FX); a.lda = a.lda2 = 64; a.K = a.K2 = 64;
    a.W = US(WM_T2I); a.W2 = US(WM_T2X); a.ntot = 4; a.bias = FP(BSUM_TAB);
    a.gamma = F(P_NG_TAB); a.beta = F(P_NG_TAB + 1); a.res = FP(FT); a.ldres = 64;
    a.outF = FP(COMB) + 0; a.ldout = 288;
    wmma_gemm<EPI_MHA, 4, 4, true, false><<<G, T, 0, stream>>>(a); }
  { GemmArgs a{}; a.A = FP(FT); a.A2 = FP(FX); a.lda = a.lda2 = 64; a.K = a.K2 = 64;
    a.W = US(WM_I2T); a.W2 = US(WM_I2X); a.ntot = 4; a.bias = FP(BSUM_IMG);
    a.gamma = F(P_NG_IMG); a.beta = F(P_NG_IMG + 1); a.res = FP(FI); a.ldres = 64;
    a.outF = FP(COMB) + 64; a.ldout = 288;
    wmma_gemm<EPI_MHA, 4, 4, true, false><<<G, T, 0, stream>>>(a); }
  { GemmArgs a{}; a.A = FP(FT); a.A2 = FP(FI); a.lda = a.lda2 = 64; a.K = a.K2 = 64;
    a.W = US(WM_X2T); a.W2 = US(WM_X2I); a.ntot = 4; a.bias = FP(BSUM_TXT);
    a.gamma = F(P_NG_TXT); a.beta = F(P_NG_TXT + 1); a.res = FP(FX); a.ldres = 64;
    a.outF = FP(COMB) + 128; a.ldout = 288;
    wmma_gemm<EPI_MHA, 4, 4, true, false><<<G, T, 0, stream>>>(a); }

  // ---- bilinear interactions (chunked Y GEMM + rank reduce) ----
  struct BL { int aoff, x2off, outoff; size_t wr; int pb, pg, pbe; };
  BL bl[3] = {
    {0,  64,  192, WBL_TI, P_BL_TI + 1, P_BL_TI + 2, P_BL_TI + 3},
    {0,  128, 224, WBL_TX, P_BL_TX + 1, P_BL_TX + 2, P_BL_TX + 3},
    {64, 128, 256, WBL_IX, P_BL_IX + 1, P_BL_IX + 2, P_BL_IX + 3},
  };
  for (int b = 0; b < 3; ++b) {
    for (int ch = 0; ch < B / CHUNK; ++ch) {
      GemmArgs a{};
      a.A = FP(COMB) + (size_t)ch * CHUNK * 288 + bl[b].aoff; a.lda = 288; a.K = 64;
      a.W = US(bl[b].wr); a.ntot = 128;
      a.outB = US(YC); a.ldoutB = 2048;
      wmma_gemm<EPI_Y, 8, 0, false, false><<<dim3(CHUNK / 128, 16), T, 0, stream>>>(a);
      k_bilin_reduce<<<dim3(64), T, 0, stream>>>(US(YC), CHUNK,
          FP(COMB) + (size_t)ch * CHUNK * 288, bl[b].x2off, bl[b].outoff,
          F(bl[b].pb), F(bl[b].pg), F(bl[b].pbe));
    }
  }

  // ---- gate, fusion head ----
  k_gate<<<dim3(256), T, 0, stream>>>(FP(COMB), F(P_WG), F(P_BG), FP(GATE), B);
  { GemmArgs a{}; a.A = FP(COMB); a.lda = 288; a.K = 288;
    a.W = US(WF1B); a.ntot = 8; a.bias = F(P_BF1);
    a.gamma = F(P_NGF); a.beta = F(P_NBF); a.gate = FP(GATE);
    a.outF = FP(H128); a.ldout = 128;
    wmma_gemm<EPI_GELU_LN, 8, 8, false, true><<<G, T, 0, stream>>>(a); }
  { GemmArgs a{}; a.A = FP(H128); a.lda = 128; a.K = 128;
    a.W = US(WC_RBF); a.ntot = 8; a.bias = FP(BC_RBF);
    a.gamma = F(P_RBF + 4); a.beta = F(P_RBF + 5);
    a.outF = FP(H64); a.ldout = 64; a.outR = FP(R64); a.ldoutR = 64;
    wmma_gemm<EPI_RB1, 8, 4, false, false><<<G, T, 0, stream>>>(a); }
  { GemmArgs a{}; a.A = FP(H64); a.lda = 64; a.K = 64;
    a.W = US(W2_RBF); a.ntot = 4; a.bias = F(P_RBF + 3);
    a.gamma = F(P_RBF + 6); a.beta = F(P_RBF + 7);
    a.res = FP(R64); a.ldres = 64; a.outF = FP(F128); a.ldout = 64;
    wmma_gemm<EPI_RB2, 4, 4, false, false><<<G, T, 0, stream>>>(a); }
  k_final<<<dim3(256), T, 0, stream>>>(FP(F128), F(P_WF2), F(P_BF2), (float*)d_out, B);
}
